// IPAttnProcessor_9904194585111
// MI455X (gfx1250) — matrix operand order repaired and checked by reading, not compiled, not run on hardware
//
#include <hip/hip_runtime.h>
#include <hip/hip_bf16.h>

typedef __attribute__((ext_vector_type(16))) __bf16 v16bf;
typedef __attribute__((ext_vector_type(8)))  float  v8f;

#define WMMA_BF16(a, b, c) \
  __builtin_amdgcn_wmma_f32_16x16x32_bf16(false, (a), false, (b), (short)0, (c), false, false)

__device__ inline v8f zero_v8f() {
  v8f z = {0.f, 0.f, 0.f, 0.f, 0.f, 0.f, 0.f, 0.f};
  return z;
}

// A fragment: 16x32 bf16 (M x K), row-major source with row stride lda (elements).
// Lane L (L<16): row M=L, elements K = [0..15]; L>=16: row M=L-16, K = [16..31].
// Per-lane bytes: one 16B-aligned 32-byte run -> 2x b128 loads.
__device__ inline v16bf load_frag_a(const __bf16* __restrict__ a, int lda) {
  const int lane = threadIdx.x & 31;
  const int hf = lane >> 4;
  const int m = lane & 15;
  v16bf f;
#pragma unroll
  for (int i = 0; i < 16; ++i) {
    const int k = (hf << 4) + i;
    f[i] = a[m * lda + k];
  }
  return f;
}

// B fragment: 32x16 bf16 (K x N), row-major source with row stride ldb.
// Lanes 0-15 hold K=0..15 (col N=lane), lanes 16-31 hold K=16..31. (strided gather)
__device__ inline v16bf load_frag_b(const __bf16* __restrict__ b, int ldb) {
  const int lane = threadIdx.x & 31;
  const int hf = lane >> 4;
  const int n = lane & 15;
  v16bf f;
#pragma unroll
  for (int i = 0; i < 16; ++i) {
    const int k = (hf << 4) + i;
    f[i] = b[k * ldb + n];
  }
  return f;
}

// B fragment from pre-transposed LDS tile BsT[n][k] with padded row stride
// LDSB_STRIDE (elements). Per-lane: 16 contiguous bf16 (32B, 16B-aligned)
// -> 2x ds_load_b128, no strided gather.
#define LDSB_STRIDE 40
__device__ inline v16bf load_frag_b_sw(const __bf16* __restrict__ bt) {
  const int lane = threadIdx.x & 31;
  const int hf = lane >> 4;
  const int n = lane & 15;
  v16bf f;
#pragma unroll
  for (int i = 0; i < 16; ++i) f[i] = bt[n * LDSB_STRIDE + (hf << 4) + i];
  return f;
}

// ---------------------------------------------------------------------------
// Tiled GEMM:  C[M x N] = A[M x K] * B[K x N]  (+ bias)
// 256 threads = 8 waves, 128x128 block tile; each wave owns 32x64 (2x4 WMMA
// tiles) -> 8 v_wmma per K-step from 6 fragment loads. A staged row-major
// (fp32/bf16 -> bf16), B staged transposed+padded for vector LDS reads.
// Rows >= M compute exact zeros; rows < storeM written (zero-pads K/V tokens).
// ---------------------------------------------------------------------------
template <typename AT, typename CT, bool BIAS>
__global__ __launch_bounds__(256) void gemm_wmma(
    const AT* __restrict__ A, const float* __restrict__ B, CT* __restrict__ C,
    const float* __restrict__ bias, int M, int N, int K, int lda, int ldb,
    int ldc, long long strideA, long long strideC, int storeM) {
  __shared__ __bf16 As[128 * 32];            // 8 KB
  __shared__ __bf16 BsT[128 * LDSB_STRIDE];  // 10 KB, BsT[n][k]

  const int tid = threadIdx.x;
  const int wave = tid >> 5;
  const int lane = tid & 31;
  const int hf = (lane >> 4) & 1;
  const int wm = (wave >> 1) * 32;  // 4 waves in M
  const int wn = (wave & 1) * 64;   // 2 waves in N
  const int mbase = blockIdx.y * 128;
  const int nbase = blockIdx.x * 128;

  const AT* Ab = A + (long long)blockIdx.z * strideA;
  CT* Cb = C + (long long)blockIdx.z * strideC;

  v8f acc[2][4];
#pragma unroll
  for (int i = 0; i < 2; ++i)
#pragma unroll
    for (int j = 0; j < 4; ++j) acc[i][j] = zero_v8f();

  for (int k0 = 0; k0 < K; k0 += 32) {
    // Stage A tile (128x32) -> bf16, zero-fill rows >= M. Coalesced reads.
#pragma unroll
    for (int it = 0; it < 16; ++it) {
      const int idx = tid + it * 256;
      const int m = idx >> 5, k = idx & 31;
      const int gm = mbase + m;
      float v = (gm < M) ? (float)Ab[(long long)gm * lda + (k0 + k)] : 0.0f;
      As[idx] = (__bf16)v;
    }
    // Stage B tile (32x128) transposed: BsT[n][k]. Coalesced global reads.
#pragma unroll
    for (int it = 0; it < 16; ++it) {
      const int idx = tid + it * 256;
      const int k = idx >> 7, n = idx & 127;
      BsT[n * LDSB_STRIDE + k] = (__bf16)B[(long long)(k0 + k) * ldb + (nbase + n)];
    }
    // Prefetch next K-tile into cache while WMMAs run (global_prefetch_b8).
    if (k0 + 32 < K) {
      const int pm = mbase + (tid >> 1);
      const int pk = (tid & 1) * 16;
      if (pm < M) __builtin_prefetch(&Ab[(long long)pm * lda + (k0 + 32) + pk], 0, 1);
      const int pn = nbase + (tid & 127);
      const int pkb = (tid >> 7) * 16;
      __builtin_prefetch(&B[(long long)(k0 + 32 + pkb) * ldb + pn], 0, 1);
    }
    __syncthreads();

    v16bf af[2], bfr[4];
#pragma unroll
    for (int s = 0; s < 2; ++s) af[s] = load_frag_a(&As[(wm + s * 16) * 32], 32);
#pragma unroll
    for (int s = 0; s < 4; ++s) bfr[s] = load_frag_b_sw(&BsT[(wn + s * 16) * LDSB_STRIDE]);
#pragma unroll
    for (int i = 0; i < 2; ++i)
#pragma unroll
      for (int j = 0; j < 4; ++j) acc[i][j] = WMMA_BF16(af[i], bfr[j], acc[i][j]);
    __syncthreads();
  }

  // Store C tile. C layout: vgpr r -> row (r + 8*hf), col = lane&15.
#pragma unroll
  for (int i = 0; i < 2; ++i) {
#pragma unroll
    for (int j = 0; j < 4; ++j) {
      const int coln = nbase + wn + j * 16 + (lane & 15);
#pragma unroll
      for (int r = 0; r < 8; ++r) {
        const int grow = mbase + wm + i * 16 + r + 8 * hf;
        if (grow < storeM && coln < N) {
          float v = acc[i][j][r];
          if (BIAS) v += bias[coln];
          Cb[(long long)grow * ldc + coln] = (CT)v;
        }
      }
    }
  }
}

// ---------------------------------------------------------------------------
// Fused dual-stream attention (main 77-key softmax + 4-key IP softmax, summed).
// Grid: (S/64, HEADS, B); 128 threads = 4 waves; each wave owns 16 queries.
// Q.K^T and P.V both via WMMA; probs transposed through LDS between them.
// ---------------------------------------------------------------------------
__global__ __launch_bounds__(128) void attn_wmma(
    const __bf16* __restrict__ Qb, const __bf16* __restrict__ Kb,
    const __bf16* __restrict__ Vb, const __bf16* __restrict__ Kip,
    const __bf16* __restrict__ Vip, __bf16* __restrict__ Ob) {
  __shared__ __bf16 Plds[4 * 16 * 96];

  const int tid = threadIdx.x;
  const int wave = tid >> 5;
  const int lane = tid & 31;
  const int hf = (lane >> 4) & 1;
  const int b = blockIdx.z;
  const int h = blockIdx.y;
  const int q0 = blockIdx.x * 64 + wave * 16;
  const float scale = 0.125f;  // 1/sqrt(64)

  const long long qrow = (long long)b * 4096 + q0;
  const __bf16* Qp = Qb + qrow * 1280 + h * 64;
  const __bf16* Kp = Kb + (long long)b * 96 * 1280 + h * 64;
  const __bf16* Vp = Vb + (long long)b * 96 * 1280 + h * 64;
  const __bf16* Kipp = Kip + (long long)b * 32 * 1280 + h * 64;
  const __bf16* Vipp = Vip + (long long)b * 32 * 1280 + h * 64;
  __bf16* Pw = &Plds[wave * 16 * 96];

  // Q fragments: 16x64 = two 16x32 A fragments.
  v16bf qf[2];
  qf[0] = load_frag_a(Qp, 1280);
  qf[1] = load_frag_a(Qp + 32, 1280);

  // ---- main stream scores: 5 key tiles of 16 (80 padded keys) ----
  v8f sc[5];
#pragma unroll
  for (int t = 0; t < 5; ++t) sc[t] = zero_v8f();
#pragma unroll
  for (int ks = 0; ks < 2; ++ks) {
#pragma unroll
    for (int t = 0; t < 5; ++t) {
      // B operand = K^T: element (d,key) = K[key][d] -> A-pattern gather.
      v16bf kf = load_frag_a(Kp + (long long)(t * 16) * 1280 + ks * 32, 1280);
      sc[t] = WMMA_BF16(qf[ks], kf, sc[t]);
    }
  }

  // scale + mask + row softmax (row = r + 8*hf, keys over 16 lanes x 5 tiles)
  float rmax[8], rsum[8];
#pragma unroll
  for (int r = 0; r < 8; ++r) rmax[r] = -3.0e38f;
#pragma unroll
  for (int t = 0; t < 5; ++t) {
    const int key = t * 16 + (lane & 15);
#pragma unroll
    for (int r = 0; r < 8; ++r) {
      float s = (key < 77) ? sc[t][r] * scale : -1.0e30f;
      sc[t][r] = s;
      rmax[r] = fmaxf(rmax[r], s);
    }
  }
#pragma unroll
  for (int off = 8; off >= 1; off >>= 1)
#pragma unroll
    for (int r = 0; r < 8; ++r) rmax[r] = fmaxf(rmax[r], __shfl_xor(rmax[r], off, 32));
#pragma unroll
  for (int r = 0; r < 8; ++r) rsum[r] = 0.f;
#pragma unroll
  for (int t = 0; t < 5; ++t)
#pragma unroll
    for (int r = 0; r < 8; ++r) {
      float p = __expf(sc[t][r] - rmax[r]);
      sc[t][r] = p;
      rsum[r] += p;
    }
#pragma unroll
  for (int off = 8; off >= 1; off >>= 1)
#pragma unroll
    for (int r = 0; r < 8; ++r) rsum[r] += __shfl_xor(rsum[r], off, 32);
  float rinv[8];
#pragma unroll
  for (int r = 0; r < 8; ++r) rinv[r] = 1.0f / rsum[r];

  // probs -> LDS (16 x 96, cols 80..95 zero) for re-fragmenting as A operand
#pragma unroll
  for (int t = 0; t < 6; ++t)
#pragma unroll
    for (int r = 0; r < 8; ++r) {
      float p = (t < 5) ? sc[t][r] * rinv[r] : 0.0f;
      Pw[(r + 8 * hf) * 96 + t * 16 + (lane & 15)] = (__bf16)p;
    }
  __syncthreads();

  // out1 = P(16x96) @ V(96x64): 3 K-steps x 4 N-tiles
  v8f o[4];
#pragma unroll
  for (int nt = 0; nt < 4; ++nt) o[nt] = zero_v8f();
#pragma unroll
  for (int ks = 0; ks < 3; ++ks) {
    v16bf pf = load_frag_a(Pw + ks * 32, 96);
#pragma unroll
    for (int nt = 0; nt < 4; ++nt) {
      v16bf vf = load_frag_b(Vp + (long long)(ks * 32) * 1280 + nt * 16, 1280);
      o[nt] = WMMA_BF16(pf, vf, o[nt]);
    }
  }

  // ---- IP stream: one 16-key tile (4 valid) ----
  v8f sip = zero_v8f();
#pragma unroll
  for (int ks = 0; ks < 2; ++ks) {
    v16bf kf = load_frag_a(Kipp + ks * 32, 1280);
    sip = WMMA_BF16(qf[ks], kf, sip);
  }
  {
    const int key = lane & 15;
    float m2[8], s2[8];
#pragma unroll
    for (int r = 0; r < 8; ++r) {
      float s = (key < 4) ? sip[r] * scale : -1.0e30f;
      sip[r] = s;
      m2[r] = s;
    }
#pragma unroll
    for (int off = 8; off >= 1; off >>= 1)
#pragma unroll
      for (int r = 0; r < 8; ++r) m2[r] = fmaxf(m2[r], __shfl_xor(m2[r], off, 32));
#pragma unroll
    for (int r = 0; r < 8; ++r) {
      float p = __expf(sip[r] - m2[r]);
      sip[r] = p;
      s2[r] = p;
    }
#pragma unroll
    for (int off = 8; off >= 1; off >>= 1)
#pragma unroll
      for (int r = 0; r < 8; ++r) s2[r] += __shfl_xor(s2[r], off, 32);
#pragma unroll
    for (int r = 0; r < 8; ++r) sip[r] = sip[r] / s2[r];
  }

  __syncthreads();
#pragma unroll
  for (int r = 0; r < 8; ++r) {
    Pw[(r + 8 * hf) * 96 + (lane & 15)] = (__bf16)sip[r];
    Pw[(r + 8 * hf) * 96 + 16 + (lane & 15)] = (__bf16)0.0f;
  }
  __syncthreads();

  {
    v16bf pf = load_frag_a(Pw, 96);
#pragma unroll
    for (int nt = 0; nt < 4; ++nt) {
      v16bf vf = load_frag_b(Vipp + nt * 16, 1280);
      o[nt] = WMMA_BF16(pf, vf, o[nt]);  // out += IP_SCALE(=1) * out_ip
    }
  }

  // store attn output (bf16) at (b, s, h, d)
#pragma unroll
  for (int nt = 0; nt < 4; ++nt) {
    const int col = h * 64 + nt * 16 + (lane & 15);
#pragma unroll
    for (int r = 0; r < 8; ++r)
      Ob[(qrow + r + 8 * hf) * 1280 + col] = (__bf16)o[nt][r];
  }
}

// ---------------------------------------------------------------------------
extern "C" void kernel_launch(void* const* d_in, const int* in_sizes, int n_in,
                              void* d_out, int out_size, void* d_ws,
                              size_t ws_size, hipStream_t stream) {
  (void)in_sizes; (void)n_in; (void)out_size; (void)ws_size;
  const float* HS   = (const float*)d_in[0];  // (8, 4096, 1280)
  const float* EHS  = (const float*)d_in[1];  // (8, 81, 2048)
  const float* Wq   = (const float*)d_in[2];  // (1280, 1280)
  const float* Wk   = (const float*)d_in[3];  // (2048, 1280)
  const float* Wv   = (const float*)d_in[4];
  const float* Wkip = (const float*)d_in[5];
  const float* Wvip = (const float*)d_in[6];
  const float* Wo   = (const float*)d_in[7];  // (1280, 1280)
  const float* bo   = (const float*)d_in[8];  // (1280,)
  float* out = (float*)d_out;                 // (8, 4096, 1280)

  char* ws = (char*)d_ws;
  const size_t szQ = 32768ull * 1280 * 2;   // Q bf16
  const size_t szK = 8ull * 96 * 1280 * 2;  // K/V bf16, 96 padded tokens
  const size_t szI = 8ull * 32 * 1280 * 2;  // K_ip/V_ip bf16, 32 padded tokens
  __bf16* Qb  = (__bf16*)(ws);
  __bf16* Kb  = (__bf16*)(ws + szQ);
  __bf16* Vb  = (__bf16*)(ws + szQ + szK);
  __bf16* Kip = (__bf16*)(ws + szQ + 2 * szK);
  __bf16* Vip = (__bf16*)(ws + szQ + 2 * szK + szI);
  __bf16* Ob  = (__bf16*)(ws + szQ + 2 * szK + 2 * szI);

  const dim3 gblk(256);

  // Q = HS @ Wq  (32768 x 1280 x 1280) -> bf16
  gemm_wmma<float, __bf16, false><<<dim3(10, 256, 1), gblk, 0, stream>>>(
      HS, Wq, Qb, nullptr, 32768, 1280, 1280, 1280, 1280, 1280, 0, 0, 32768);

  // K/V = ctx @ Wk/Wv per batch (77 x 1280 x 2048), zero-padded to 96 rows
  gemm_wmma<float, __bf16, false><<<dim3(10, 1, 8), gblk, 0, stream>>>(
      EHS, Wk, Kb, nullptr, 77, 1280, 2048, 2048, 1280, 1280,
      81ll * 2048, 96ll * 1280, 96);
  gemm_wmma<float, __bf16, false><<<dim3(10, 1, 8), gblk, 0, stream>>>(
      EHS, Wv, Vb, nullptr, 77, 1280, 2048, 2048, 1280, 1280,
      81ll * 2048, 96ll * 1280, 96);

  // K_ip/V_ip = ip_ctx @ Wk_ip/Wv_ip (4 x 1280 x 2048), zero-padded to 32 rows
  gemm_wmma<float, __bf16, false><<<dim3(10, 1, 8), gblk, 0, stream>>>(
      EHS + 77ll * 2048, Wkip, Kip, nullptr, 4, 1280, 2048, 2048, 1280, 1280,
      81ll * 2048, 32ll * 1280, 32);
  gemm_wmma<float, __bf16, false><<<dim3(10, 1, 8), gblk, 0, stream>>>(
      EHS + 77ll * 2048, Wvip, Vip, nullptr, 4, 1280, 2048, 2048, 1280, 1280,
      81ll * 2048, 32ll * 1280, 32);

  // fused dual-softmax attention
  attn_wmma<<<dim3(64, 20, 8), dim3(128), 0, stream>>>(Qb, Kb, Vb, Kip, Vip, Ob);

  // out = attnOut @ Wo + bo  -> fp32
  gemm_wmma<__bf16, float, true><<<dim3(10, 256, 1), gblk, 0, stream>>>(
      Ob, Wo, out, bo, 32768, 1280, 1280, 1280, 1280, 1280, 0, 0, 32768);
}
